// VishwamAIModel_63067299774471
// MI455X (gfx1250) — compile-verified, matrix operand order checked by reading, not run on hardware
//
#include <hip/hip_runtime.h>
#include <hip/hip_bf16.h>

// ---------------------------------------------------------------------------
// Model dims (VishwamAI MoE-transformer reference)
// ---------------------------------------------------------------------------
#define VOC 50257
#define DM  512
#define FF  2048
#define NH  8
#define HD  64          // DM / NH
#define NE  4
#define HID 128
#define BB  4
#define SS  1024
#define NTOK (BB * SS)  // 4096

typedef __attribute__((ext_vector_type(16))) __bf16 v16bf;
typedef __attribute__((ext_vector_type(8)))  float  v8f;

union FragBF { v16bf v; unsigned int u[8]; };

__device__ __forceinline__ unsigned short f2bf(float f) {
    unsigned int u = __float_as_uint(f);
    u += 0x7FFFu + ((u >> 16) & 1u);      // round-to-nearest-even
    return (unsigned short)(u >> 16);
}
// RNE-round both values on full 32-bit regs, then fuse the pack into a single
// v_perm_b32: result bytes = {hi[3],hi[2],lo[3],lo[2]}  (sel 0x07060302).
__device__ __forceinline__ unsigned int pk2bf(float lo, float hi) {
    unsigned int ul = __float_as_uint(lo);
    unsigned int uh = __float_as_uint(hi);
    ul += 0x7FFFu + ((ul >> 16) & 1u);
    uh += 0x7FFFu + ((uh >> 16) & 1u);
    return __builtin_amdgcn_perm(uh, ul, 0x07060302u);
}

// ---------------------------------------------------------------------------
// Batched GEMM: C[z] = alpha * A[z] x op(B[z]) (+ bias) (+ beta*C[z])
// fp32 in global; bf16 conversion while staging to LDS; WMMA bf16 MACs.
// Block tile 128(M) x 128(N) x 32(K); 8 waves in 4x2, each 32x64 = 2x4 WMMA
// tiles -> 8 v_wmma per wave per K-step.
// Software pipeline: register-staged global loads + double-buffered LDS;
// next tile's global_load_b128 issued before the current tile's WMMA burst.
// REQUIREMENTS (all call sites satisfy): K % 32 == 0, M % 128 == 0.
// Template: TRANSB (B indexed [n][k] vs [k][n]), NFULL (N % 128 == 0).
// ---------------------------------------------------------------------------
#define LDP 34   // padded LDS row stride (ushorts) = 17 dwords -> conflict-free

template <int TRANSB, int NFULL>
__global__ __launch_bounds__(256) void gemm_bf16_wmma(
    const float* __restrict__ A, long long strideA, int lda,
    const float* __restrict__ B, long long strideB, int ldb,
    float* __restrict__ C, long long strideC, int ldc,
    const float* __restrict__ bias,
    int M, int N, int K, float alpha, int beta)
{
    __shared__ unsigned short As[2][128 * LDP];
    __shared__ unsigned short Bs[2][128 * LDP];

    const int tid  = threadIdx.x;
    const int lane = tid & 31;
    const int wid  = tid >> 5;
    const int wm   = (wid & 3) * 32;       // wave M offset in block tile
    const int wn   = (wid >> 2) * 64;      // wave N offset in block tile
    const int bm   = blockIdx.y * 128;
    const int bn   = blockIdx.x * 128;

    A += (long long)blockIdx.z * strideA;
    B += (long long)blockIdx.z * strideB;
    C += (long long)blockIdx.z * strideC;

    v8f acc[2][4];
#pragma unroll
    for (int mi = 0; mi < 2; ++mi)
#pragma unroll
        for (int ni = 0; ni < 4; ++ni)
#pragma unroll
            for (int r = 0; r < 8; ++r) acc[mi][ni][r] = 0.0f;

    const int mloc = lane & 15;   // M (or N) index within a 16x16 tile
    const int half = lane >> 4;   // lane group 0/1 selects K sub-range

    // ---- loop-invariant global pointers (strength-reduced per K-step)
    const float* pa = A + (long long)(bm + (tid >> 3)) * lda + ((tid & 7) << 2);
    const long long aoff = (long long)32 * lda;      // +32 rows between chunks

    const float* pb = nullptr;
    long long boff = 0, badv = 0;
    if (NFULL) {
        if (TRANSB) {
            pb = B + (long long)(bn + (tid >> 3)) * ldb + ((tid & 7) << 2);
            boff = (long long)32 * ldb;   // +32 n-rows between chunks
            badv = 32;                    // advance 32 k per step
        } else {
            pb = B + (long long)(tid >> 5) * ldb + bn + ((tid & 31) << 2);
            boff = (long long)8 * ldb;    // +8 k-rows between chunks
            badv = (long long)32 * ldb;   // advance 32 k-rows per step
        }
    }

    // ---- register staging
    float4 ra[4];
    float4 rb4[4];
    float  rbs[16];

    auto loadA = [&]() {
#pragma unroll
        for (int ii = 0; ii < 4; ++ii) ra[ii] = *(const float4*)(pa + ii * aoff);
        pa += 32;
    };
    auto loadB = [&](int kc) {
        if (NFULL) {
#pragma unroll
            for (int ii = 0; ii < 4; ++ii) rb4[ii] = *(const float4*)(pb + ii * boff);
            pb += badv;
        } else {
#pragma unroll
            for (int ii = 0; ii < 16; ++ii) {
                int n = (tid >> 5) + ii * 8, kk = tid & 31;
                int gn = bn + n;
                rbs[ii] = (gn < N)
                    ? (TRANSB ? B[(long long)gn * ldb + kc + kk]
                              : B[(long long)(kc + kk) * ldb + gn])
                    : 0.0f;
            }
        }
    };
    const int sa_r = tid >> 3, sa_k = (tid & 7) << 2;
    auto storeA = [&](int buf) {
#pragma unroll
        for (int ii = 0; ii < 4; ++ii) {
            unsigned short* dst = &As[buf][(sa_r + ii * 32) * LDP + sa_k];
            *(unsigned int*)(dst)     = pk2bf(ra[ii].x, ra[ii].y);
            *(unsigned int*)(dst + 2) = pk2bf(ra[ii].z, ra[ii].w);
        }
    };
    auto storeB = [&](int buf) {
        if (NFULL) {
            if (TRANSB) {
#pragma unroll
                for (int ii = 0; ii < 4; ++ii) {
                    unsigned short* dst = &Bs[buf][(sa_r + ii * 32) * LDP + sa_k];
                    *(unsigned int*)(dst)     = pk2bf(rb4[ii].x, rb4[ii].y);
                    *(unsigned int*)(dst + 2) = pk2bf(rb4[ii].z, rb4[ii].w);
                }
            } else {
                const int n0 = (tid & 31) << 2;
#pragma unroll
                for (int ii = 0; ii < 4; ++ii) {
                    int kk = (tid >> 5) + ii * 8;
                    Bs[buf][(n0    ) * LDP + kk] = f2bf(rb4[ii].x);
                    Bs[buf][(n0 + 1) * LDP + kk] = f2bf(rb4[ii].y);
                    Bs[buf][(n0 + 2) * LDP + kk] = f2bf(rb4[ii].z);
                    Bs[buf][(n0 + 3) * LDP + kk] = f2bf(rb4[ii].w);
                }
            }
        } else {
#pragma unroll
            for (int ii = 0; ii < 16; ++ii) {
                int n = (tid >> 5) + ii * 8, kk = tid & 31;
                Bs[buf][n * LDP + kk] = f2bf(rbs[ii]);
            }
        }
    };

    // ---- prologue: fetch tile 0 into registers
    loadA();
    loadB(0);

    int buf = 0;
    for (int k0 = 0; k0 < K; k0 += 32) {
        storeA(buf);                   // s_wait_loadcnt happens here (convert)
        storeB(buf);
        __syncthreads();               // tile `buf` visible to all waves
        if (k0 + 32 < K) {             // issue NEXT tile's global loads now;
            loadA();                   // they fly during the WMMA burst below
            loadB(k0 + 32);
            if (k0 + 64 < K) __builtin_prefetch(pa, 0, 2);  // tile+2 into L2
        }

        // ---- fragments per ISA 16-bit 16x32 A layout:
        //   vgpr v holds K pair k = 2*(v&3) + 8*half + 16*(v>=4)
        FragBF af[2], bf[4];
#pragma unroll
        for (int mi = 0; mi < 2; ++mi) {
            const unsigned short* src = &As[buf][(wm + mi * 16 + mloc) * LDP];
#pragma unroll
            for (int v = 0; v < 8; ++v) {
                int kk = ((v & 3) << 1) + (half << 3) + ((v >> 2) << 4);
                af[mi].u[v] = *(const unsigned int*)&src[kk];
            }
        }
#pragma unroll
        for (int ni = 0; ni < 4; ++ni) {
            const unsigned short* src = &Bs[buf][(wn + ni * 16 + mloc) * LDP];
#pragma unroll
            for (int v = 0; v < 8; ++v) {
                int kk = ((v & 3) << 1) + (half << 3) + ((v >> 2) << 4);
                bf[ni].u[v] = *(const unsigned int*)&src[kk];
            }
        }
#pragma unroll
        for (int mi = 0; mi < 2; ++mi)
#pragma unroll
            for (int ni = 0; ni < 4; ++ni)
                acc[mi][ni] = __builtin_amdgcn_wmma_f32_16x16x32_bf16(
                    false, af[mi].v, false, bf[ni].v,
                    (short)0, acc[mi][ni], false, false);
        buf ^= 1;
    }

    // ---- store: C/D layout -> vgpr r is row (half*8 + r), lane&15 is col
    //      M % 128 == 0 guaranteed, so no row guard needed.
#pragma unroll
    for (int mi = 0; mi < 2; ++mi)
#pragma unroll
        for (int ni = 0; ni < 4; ++ni) {
            int col = bn + wn + ni * 16 + mloc;
            if (NFULL || col < N) {
#pragma unroll
                for (int r = 0; r < 8; ++r) {
                    int row = bm + wm + mi * 16 + half * 8 + r;
                    float v = alpha * acc[mi][ni][r];
                    if (bias) v += bias[col];
                    long long off = (long long)row * ldc + col;
                    if (beta) v += C[off];
                    C[off] = v;
                }
            }
        }
}

// ---------------------------------------------------------------------------
// Embedding gather (optionally masked by expert route: ids->0 if not routed)
// ---------------------------------------------------------------------------
__global__ void gather_embed(const int* __restrict__ ids,
                             const int* __restrict__ route, int expert,
                             const float* __restrict__ emb,
                             float* __restrict__ x)
{
    int i = blockIdx.x * 256 + threadIdx.x;     // over NTOK*DM/4 = 2^19
    if (i >= NTOK * (DM / 4)) return;
    int d = i & (DM / 4 - 1);
    int t = i >> 7;
    int tok = ids[t];
    if (route && route[t] != expert) tok = 0;
    ((float4*)x)[i] = ((const float4*)emb)[(long long)tok * (DM / 4) + d];
}

// [B,S,H*hd] -> [B*H, S, hd], float4-wide
__global__ void pack_heads(const float* __restrict__ x, float* __restrict__ y)
{
    int i = blockIdx.x * 256 + threadIdx.x;     // over NTOK*DM/4
    if (i >= NTOK * (DM / 4)) return;
    int j = i & 15, h = (i >> 4) & 7, s = (i >> 7) & 1023, b = i >> 17;
    ((float4*)y)[(((b * NH + h) * SS) + s) * (HD / 4) + j] = ((const float4*)x)[i];
}

// [B*H, S, hd] -> [B,S,H*hd], float4-wide
__global__ void unpack_heads(const float* __restrict__ y, float* __restrict__ x)
{
    int i = blockIdx.x * 256 + threadIdx.x;
    if (i >= NTOK * (DM / 4)) return;
    int j = i & 15, s = (i >> 4) & 1023, bh = i >> 14;
    int b = bh >> 3, h = bh & 7;
    ((float4*)x)[((b * SS + s) * (DM / 4)) + h * (HD / 4) + j] = ((const float4*)y)[i];
}

// Row softmax, specialized to 1024 columns: one float4 per thread.
__global__ __launch_bounds__(256) void softmax_rows_1k(float* __restrict__ s)
{
    __shared__ float red[256];
    float4* p = (float4*)(s + (long long)blockIdx.x * SS);
    const int tid = threadIdx.x;
    float4 v = p[tid];
    float m = fmaxf(fmaxf(v.x, v.y), fmaxf(v.z, v.w));
    red[tid] = m; __syncthreads();
    for (int o = 128; o > 0; o >>= 1) {
        if (tid < o) red[tid] = fmaxf(red[tid], red[tid + o]);
        __syncthreads();
    }
    m = red[0]; __syncthreads();
    v.x = __expf(v.x - m); v.y = __expf(v.y - m);
    v.z = __expf(v.z - m); v.w = __expf(v.w - m);
    red[tid] = v.x + v.y + v.z + v.w; __syncthreads();
    for (int o = 128; o > 0; o >>= 1) {
        if (tid < o) red[tid] += red[tid + o];
        __syncthreads();
    }
    float inv = 1.0f / red[0];
    v.x *= inv; v.y *= inv; v.z *= inv; v.w *= inv;
    p[tid] = v;
}

// gate = x @ Wg + bg ; idx = argmax_e (first-max tie break like jnp.argmax)
__global__ void gate_argmax(const float* __restrict__ x,
                            const float* __restrict__ Wg,
                            const float* __restrict__ bg,
                            int* __restrict__ idx)
{
    int t = blockIdx.x * 256 + threadIdx.x;
    if (t >= NTOK) return;
    const float*  xr  = x + (long long)t * DM;
    const float4* wg4 = (const float4*)Wg;
    float4 a = *(const float4*)bg;
    for (int d = 0; d < DM; ++d) {
        float xv = xr[d];
        float4 w = wg4[d];
        a.x += xv * w.x; a.y += xv * w.y; a.z += xv * w.z; a.w += xv * w.w;
    }
    float best = a.x; int bi = 0;
    if (a.y > best) { best = a.y; bi = 1; }
    if (a.z > best) { best = a.z; bi = 2; }
    if (a.w > best) { best = a.w; bi = 3; }
    idx[t] = bi;
}

__device__ __forceinline__ float sigm(float x) { return 1.0f / (1.0f + __expf(-x)); }

// haiku LSTM: gates [i,g,f,o], forget bias +1. One block, 512 threads = (b,u).
__global__ __launch_bounds__(512) void lstm_kernel(
    const float* __restrict__ x,   // [B,S,DM]
    const float* __restrict__ Wl,  // [DM+HID, 4*HID]
    const float* __restrict__ bl,  // [4*HID]
    float* __restrict__ hs)        // [B,S,HID]
{
    __shared__ float xs[BB * DM];
    __shared__ float hsh[BB * HID];
    const int tid = threadIdx.x;
    const int b = tid >> 7, u = tid & (HID - 1);
    float c = 0.0f;
    hsh[tid] = 0.0f;
    const float bi0 = bl[u], bg0 = bl[HID + u], bf0 = bl[2 * HID + u], bo0 = bl[3 * HID + u];
    for (int t = 0; t < SS; ++t) {
        for (int i = tid; i < BB * DM; i += 512)
            xs[i] = x[((long long)(i >> 9) * SS + t) * DM + (i & (DM - 1))];
        __syncthreads();
        float gi = bi0, gg = bg0, gf = bf0, go = bo0;
        const float* xb = &xs[b * DM];
        const float* hb = &hsh[b * HID];
        for (int kk = 0; kk < DM + HID; ++kk) {
            float xv = (kk < DM) ? xb[kk] : hb[kk - DM];
            const float* wr = Wl + (long long)kk * (4 * HID);
            gi += xv * wr[u];
            gg += xv * wr[HID + u];
            gf += xv * wr[2 * HID + u];
            go += xv * wr[3 * HID + u];
        }
        c = sigm(gf + 1.0f) * c + sigm(gi) * tanhf(gg);
        float h = sigm(go) * tanhf(c);
        __syncthreads();                 // all reads of old h done
        hsh[b * HID + u] = h;
        hs[((long long)b * SS + t) * HID + u] = h;
        __syncthreads();
    }
}

__global__ void head_kernel(const float* __restrict__ mem,
                            const float* __restrict__ Wd,
                            const float* __restrict__ bd,
                            float* __restrict__ out)
{
    int t = blockIdx.x * 256 + threadIdx.x;
    if (t >= NTOK) return;
    const float4* m4 = (const float4*)(mem + (long long)t * HID);
    const float4* w4 = (const float4*)Wd;
    float a = bd[0];
    for (int j = 0; j < HID / 4; ++j) {
        float4 mv = m4[j], wv = w4[j];
        a += mv.x * wv.x + mv.y * wv.y + mv.z * wv.z + mv.w * wv.w;
    }
    out[t] = a;
}

// ---------------------------------------------------------------------------
// Host orchestration
// ---------------------------------------------------------------------------
struct TrunkW {
    const float *Wq, *bq, *Wk, *bk, *Wv, *bv, *Wo, *bo, *W1, *b1, *W2, *b2;
};

extern "C" void kernel_launch(void* const* d_in, const int* in_sizes, int n_in,
                              void* d_out, int out_size, void* d_ws, size_t ws_size,
                              hipStream_t stream)
{
    (void)in_sizes; (void)n_in; (void)out_size; (void)ws_size;

    const int*   ids   = (const int*)  d_in[0];
    const float* emb_s = (const float*)d_in[1];
    TrunkW ws_shared = { (const float*)d_in[2],  (const float*)d_in[3],
                         (const float*)d_in[4],  (const float*)d_in[5],
                         (const float*)d_in[6],  (const float*)d_in[7],
                         (const float*)d_in[8],  (const float*)d_in[9],
                         (const float*)d_in[10], (const float*)d_in[11],
                         (const float*)d_in[12], (const float*)d_in[13] };
    const float* Wg    = (const float*)d_in[14];
    const float* bg    = (const float*)d_in[15];
    const float* emb_e = (const float*)d_in[16];
    const float* Wq_e  = (const float*)d_in[17]; const float* bq_e = (const float*)d_in[18];
    const float* Wk_e  = (const float*)d_in[19]; const float* bk_e = (const float*)d_in[20];
    const float* Wv_e  = (const float*)d_in[21]; const float* bv_e = (const float*)d_in[22];
    const float* Wo_e  = (const float*)d_in[23]; const float* bo_e = (const float*)d_in[24];
    const float* W1_e  = (const float*)d_in[25]; const float* b1_e = (const float*)d_in[26];
    const float* W2_e  = (const float*)d_in[27]; const float* b2_e = (const float*)d_in[28];
    TrunkW ws_comb = { (const float*)d_in[29], (const float*)d_in[30],
                       (const float*)d_in[31], (const float*)d_in[32],
                       (const float*)d_in[33], (const float*)d_in[34],
                       (const float*)d_in[35], (const float*)d_in[36],
                       (const float*)d_in[37], (const float*)d_in[38],
                       (const float*)d_in[39], (const float*)d_in[40] };
    const float* Wl = (const float*)d_in[41];
    const float* bl = (const float*)d_in[42];
    const float* Wd = (const float*)d_in[43];
    const float* bd = (const float*)d_in[44];

    // ---- workspace layout (floats)
    float* W = (float*)d_ws;
    const long long MT = (long long)NTOK * DM;       // 2,097,152
    float* xb   = W;
    float* qb   = W + 1 * MT;
    float* kb   = W + 2 * MT;
    float* vb   = W + 3 * MT;
    float* qh   = W + 4 * MT;
    float* kh   = W + 5 * MT;
    float* vh   = W + 6 * MT;
    float* oh   = W + 7 * MT;
    float* ctx  = W + 8 * MT;
    float* attb = W + 9 * MT;
    float* embB = W + 10 * MT;
    float* aggb = W + 11 * MT;
    float* memb = W + 12 * MT;                       // NTOK*HID
    int*   idxb = (int*)(W + 12 * MT + (long long)NTOK * HID);
    float* sc   = W + 13 * MT;                       // 32*1024*1024 (scores & FFN mid)

    auto gemm = [&](const float* A, long long sA, int lda,
                    const float* Bp, long long sB, int ldb, int tB,
                    float* Cp, long long sC, int ldc, const float* bias,
                    int M, int N, int K, float alpha, int beta, int batch) {
        dim3 g((unsigned)((N + 127) / 128), (unsigned)((M + 127) / 128), (unsigned)batch);
        if (tB) {
            if ((N & 127) == 0)
                gemm_bf16_wmma<1, 1><<<g, 256, 0, stream>>>(A, sA, lda, Bp, sB, ldb,
                                                            Cp, sC, ldc, bias, M, N, K, alpha, beta);
            else
                gemm_bf16_wmma<1, 0><<<g, 256, 0, stream>>>(A, sA, lda, Bp, sB, ldb,
                                                            Cp, sC, ldc, bias, M, N, K, alpha, beta);
        } else {
            if ((N & 127) == 0)
                gemm_bf16_wmma<0, 1><<<g, 256, 0, stream>>>(A, sA, lda, Bp, sB, ldb,
                                                            Cp, sC, ldc, bias, M, N, K, alpha, beta);
            else
                gemm_bf16_wmma<0, 0><<<g, 256, 0, stream>>>(A, sA, lda, Bp, sB, ldb,
                                                            Cp, sC, ldc, bias, M, N, K, alpha, beta);
        }
    };

    // One transformer block: x -> MHA -> Linear(F) -> Linear(D) -> out
    auto block_fwd = [&](const float* x, const TrunkW& w, float* out, int beta) {
        gemm(x, 0, DM, w.Wq, 0, DM, 0, qb, 0, DM, w.bq, NTOK, DM, DM, 1.f, 0, 1);
        gemm(x, 0, DM, w.Wk, 0, DM, 0, kb, 0, DM, w.bk, NTOK, DM, DM, 1.f, 0, 1);
        gemm(x, 0, DM, w.Wv, 0, DM, 0, vb, 0, DM, w.bv, NTOK, DM, DM, 1.f, 0, 1);
        pack_heads<<<2048, 256, 0, stream>>>(qb, qh);
        pack_heads<<<2048, 256, 0, stream>>>(kb, kh);
        pack_heads<<<2048, 256, 0, stream>>>(vb, vh);
        // scores[bh] = (1/8) * Q[bh] x K[bh]^T          (32 batches, 1024x1024x64)
        gemm(qh, (long long)SS * HD, HD, kh, (long long)SS * HD, HD, 1,
             sc, (long long)SS * SS, SS, nullptr, SS, SS, HD, 0.125f, 0, NH * BB);
        softmax_rows_1k<<<NH * BB * SS, 256, 0, stream>>>(sc);
        // O[bh] = P[bh] x V[bh]                          (1024x64x1024)
        gemm(sc, (long long)SS * SS, SS, vh, (long long)SS * HD, HD, 0,
             oh, (long long)SS * HD, HD, nullptr, SS, HD, SS, 1.f, 0, NH * BB);
        unpack_heads<<<2048, 256, 0, stream>>>(oh, ctx);
        gemm(ctx, 0, DM, w.Wo, 0, DM, 0, attb, 0, DM, w.bo, NTOK, DM, DM, 1.f, 0, 1);
        gemm(attb, 0, DM, w.W1, 0, FF, 0, sc, 0, FF, w.b1, NTOK, FF, DM, 1.f, 0, 1);
        gemm(sc, 0, FF, w.W2, 0, DM, 0, out, 0, DM, w.b2, NTOK, DM, FF, 1.f, beta, 1);
    };

    // 1) shared trunk
    gather_embed<<<2048, 256, 0, stream>>>(ids, nullptr, -1, emb_s, xb);
    block_fwd(xb, ws_shared, embB, 0);

    // 2) gating: per-token argmax expert
    gate_argmax<<<NTOK / 256, 256, 0, stream>>>(embB, Wg, bg, idxb);

    // 3) experts (dense dispatch with masked token ids), summed via beta
    for (int e = 0; e < NE; ++e) {
        TrunkW we = { Wq_e + (long long)e * DM * DM, bq_e + (long long)e * DM,
                      Wk_e + (long long)e * DM * DM, bk_e + (long long)e * DM,
                      Wv_e + (long long)e * DM * DM, bv_e + (long long)e * DM,
                      Wo_e + (long long)e * DM * DM, bo_e + (long long)e * DM,
                      W1_e + (long long)e * DM * FF, b1_e + (long long)e * FF,
                      W2_e + (long long)e * FF * DM, b2_e + (long long)e * DM };
        gather_embed<<<2048, 256, 0, stream>>>(ids, idxb, e,
                                               emb_e + (long long)e * VOC * DM, xb);
        block_fwd(xb, we, aggb, e > 0);
    }

    // 4) combine block (attention + FFN on aggregated expert output)
    block_fwd(aggb, ws_comb, xb, 0);

    // 5) LSTM memory network + dense head
    lstm_kernel<<<1, 512, 0, stream>>>(xb, Wl, bl, memb);
    head_kernel<<<NTOK / 256, 256, 0, stream>>>(memb, Wd, bd, (float*)d_out);
}
